// CategoricalGraphAtt_4105988735456
// MI455X (gfx1250) — compile-verified, matrix operand order checked by reading, not run on hardware
//
#include <hip/hip_runtime.h>
#include <math.h>

#define N_COMPANY 16384
#define N_SECTOR  64
#define PER_SECTOR 256
#define WIN   32
#define D_IN  16
#define HG    64
#define G3    192          // 3 * HG
#define FEAT  (WIN * D_IN) // 512

typedef __attribute__((ext_vector_type(16))) __bf16 v16bf;
typedef __attribute__((ext_vector_type(8)))  float  v8f;

union BF16x16 { v16bf v; unsigned short u[16]; };

__device__ __forceinline__ unsigned short f2bf(float f) {
    unsigned u = __float_as_uint(f);
    unsigned r = u + 0x7FFFu + ((u >> 16) & 1u); // RNE
    return (unsigned short)(r >> 16);
}

// Build a 16x32 bf16 A/B fragment from LDS row-major [16][ldk] (element (r,k) at base[r*ldk+k]).
// Per ISA 16-bit A layout: lane m=l%16; vgpr i<4: K=8*hi+2i+hf ; i>=4: K=16+8*hi+2(i-4)+hf.
__device__ __forceinline__ v16bf load_frag16(const unsigned short* base, int ldk, int lane) {
    BF16x16 r;
    int hi8 = (lane >> 4) & 1;
    const unsigned short* row = base + (size_t)(lane & 15) * ldk;
#pragma unroll
    for (int i = 0; i < 8; ++i) {
        int kb = (i < 4) ? (8 * hi8 + 2 * i) : (16 + 8 * hi8 + 2 * (i - 4));
        r.u[2 * i]     = row[kb];
        r.u[2 * i + 1] = row[kb + 1];
    }
    return r.v;
}

__device__ __forceinline__ unsigned fflip(float f) {
    unsigned b = __float_as_uint(f);
    return b ^ ((b & 0x80000000u) ? 0xFFFFFFFFu : 0x80000000u);
}
__device__ __forceinline__ float funflip(unsigned k) {
    return __uint_as_float(k ^ ((k & 0x80000000u) ? 0x80000000u : 0xFFFFFFFFu));
}

// ---------------- BatchNorm ----------------
__global__ void bn_stats_kernel(const float* __restrict__ x, float* __restrict__ mu, float* __restrict__ rs) {
    int f = blockIdx.x;            // feature f = w*16 + d
    int w = f >> 4, d = f & 15;
    const float* base = x + (size_t)w * N_COMPANY * D_IN + d;
    __shared__ float ssum[256], ssq[256];
    float s = 0.f, q = 0.f;
    for (int n = threadIdx.x; n < N_COMPANY; n += 256) {
        float v = base[(size_t)n * D_IN];
        s += v; q += v * v;
    }
    ssum[threadIdx.x] = s; ssq[threadIdx.x] = q;
    __syncthreads();
    for (int st = 128; st > 0; st >>= 1) {
        if (threadIdx.x < st) { ssum[threadIdx.x] += ssum[threadIdx.x + st]; ssq[threadIdx.x] += ssq[threadIdx.x + st]; }
        __syncthreads();
    }
    if (threadIdx.x == 0) {
        float m = ssum[0] * (1.f / N_COMPANY);
        float var = ssq[0] * (1.f / N_COMPANY) - m * m;
        mu[f] = m;
        rs[f] = rsqrtf(var + 1e-5f);
    }
}

__global__ void bn_apply_kernel(const float* __restrict__ x, const float* __restrict__ mu,
                                const float* __restrict__ rs, const float* __restrict__ g,
                                const float* __restrict__ b, unsigned short* __restrict__ xn) {
    size_t i = (size_t)blockIdx.x * blockDim.x + threadIdx.x;
    size_t total = (size_t)WIN * N_COMPANY * D_IN;
    if (i >= total) return;
    int d = (int)(i % D_IN);
    int w = (int)(i / ((size_t)N_COMPANY * D_IN));
    int f = w * D_IN + d;
    float v = (x[i] - mu[f]) * rs[f] * g[f] + b[f];
    xn[i] = f2bf(v);
}

__global__ void zero_kernel(unsigned* __restrict__ p, size_t n) {
    size_t i = (size_t)blockIdx.x * blockDim.x + threadIdx.x;
    if (i < n) p[i] = 0u;
}

// ---------------- GRU (WMMA, 1 wave / 16 companies, async x-tile copies) ----------------
__global__ __launch_bounds__(32) void gru_kernel(
    const unsigned short* __restrict__ xn,
    const float* __restrict__ Wih, const float* __restrict__ Whh,
    const float* __restrict__ bih, const float* __restrict__ bhh,
    const float* __restrict__ h0, float* __restrict__ seq)
{
    __shared__ unsigned short sWih[G3][32];   // transposed [n][k], K padded 16->32
    __shared__ unsigned short sWhh[G3][64];   // transposed [n][k]
    __shared__ unsigned short sH[16][64];     // h state bf16
    __shared__ unsigned short sX[16][32];     // x tile bf16 (K padded)
    __shared__ float sBrz[128], sBi[64], sBh[64];

    int lane = threadIdx.x;
    int n0 = blockIdx.x * 16;

    for (int idx = lane; idx < G3 * 32; idx += 32) {
        int n = idx >> 5, k = idx & 31;
        sWih[n][k] = (k < D_IN) ? f2bf(Wih[(size_t)k * G3 + n]) : (unsigned short)0;
    }
    for (int idx = lane; idx < G3 * 64; idx += 32) {
        int n = idx >> 6, k = idx & 63;
        sWhh[n][k] = f2bf(Whh[(size_t)k * G3 + n]);
    }
    for (int idx = lane; idx < 128; idx += 32) sBrz[idx] = bih[idx] + bhh[idx];
    for (int idx = lane; idx < 64; idx += 32) { sBi[idx] = bih[128 + idx]; sBh[idx] = bhh[128 + idx]; }
    for (int idx = lane; idx < 16 * 32; idx += 32) { int m = idx >> 5, k = idx & 31; if (k >= D_IN) sX[m][k] = 0; }

    int mb = (lane >> 4) * 8;   // C-layout row base
    int nl = lane & 15;         // C-layout column within tile
    float hreg[4][8];
#pragma unroll
    for (int j = 0; j < 4; ++j)
#pragma unroll
        for (int v = 0; v < 8; ++v) {
            float hv = h0[(size_t)(n0 + mb + v) * HG + j * 16 + nl];
            hreg[j][v] = hv;
            sH[mb + v][j * 16 + nl] = f2bf(hv);
        }

    // per-lane async-copy coordinates: 32 lanes x 16B cover the 16x16 bf16 tile
    int xrow = lane >> 1, xhalf = lane & 1;
    unsigned ldsXa = (unsigned)(size_t)&sX[xrow][xhalf * 8];
    const unsigned short* gx0 = xn + (size_t)(n0 + xrow) * D_IN + (size_t)xhalf * 8;
    __syncthreads();

    for (int t = 0; t < WIN; ++t) {
        // async global -> LDS copy of the 512B x-tile (no VGPR staging; ASYNCcnt)
        {
            unsigned long long ga = (unsigned long long)(size_t)(gx0 + (size_t)t * N_COMPANY * D_IN);
            asm volatile("global_load_async_to_lds_b128 %0, %1, off"
                         :: "v"(ldsXa), "v"(ga) : "memory");
            if (t + 1 < WIN)   // warm L2 for next tile
                __builtin_prefetch(gx0 + (size_t)(t + 1) * N_COMPANY * D_IN, 0, 1);
            asm volatile("s_wait_asynccnt 0x0" ::: "memory");
        }
        __syncthreads();

        v16bf ax  = load_frag16(&sX[0][0], 32, lane);
        v16bf ah0 = load_frag16(&sH[0][0], 64, lane);
        v16bf ah1 = load_frag16(&sH[0][32], 64, lane);

        v8f accR[4], accZ[4], accI[4], accN[4];
#pragma unroll
        for (int j = 0; j < 4; ++j) {
            float br = sBrz[j * 16 + nl];
            float bz = sBrz[64 + j * 16 + nl];
            float bi = sBi[j * 16 + nl];
            float bh = sBh[j * 16 + nl];
#pragma unroll
            for (int v = 0; v < 8; ++v) { accR[j][v] = br; accZ[j][v] = bz; accI[j][v] = bi; accN[j][v] = bh; }
        }
#pragma unroll
        for (int j = 0; j < 4; ++j) {
            v16bf b;
            b = load_frag16(&sWih[j * 16][0], 32, lane);
            accR[j] = __builtin_amdgcn_wmma_f32_16x16x32_bf16(false, ax,  false, b, (short)0, accR[j], false, false);
            b = load_frag16(&sWhh[j * 16][0], 64, lane);
            accR[j] = __builtin_amdgcn_wmma_f32_16x16x32_bf16(false, ah0, false, b, (short)0, accR[j], false, false);
            b = load_frag16(&sWhh[j * 16][32], 64, lane);
            accR[j] = __builtin_amdgcn_wmma_f32_16x16x32_bf16(false, ah1, false, b, (short)0, accR[j], false, false);

            b = load_frag16(&sWih[64 + j * 16][0], 32, lane);
            accZ[j] = __builtin_amdgcn_wmma_f32_16x16x32_bf16(false, ax,  false, b, (short)0, accZ[j], false, false);
            b = load_frag16(&sWhh[64 + j * 16][0], 64, lane);
            accZ[j] = __builtin_amdgcn_wmma_f32_16x16x32_bf16(false, ah0, false, b, (short)0, accZ[j], false, false);
            b = load_frag16(&sWhh[64 + j * 16][32], 64, lane);
            accZ[j] = __builtin_amdgcn_wmma_f32_16x16x32_bf16(false, ah1, false, b, (short)0, accZ[j], false, false);

            b = load_frag16(&sWih[128 + j * 16][0], 32, lane);
            accI[j] = __builtin_amdgcn_wmma_f32_16x16x32_bf16(false, ax,  false, b, (short)0, accI[j], false, false);
            b = load_frag16(&sWhh[128 + j * 16][0], 64, lane);
            accN[j] = __builtin_amdgcn_wmma_f32_16x16x32_bf16(false, ah0, false, b, (short)0, accN[j], false, false);
            b = load_frag16(&sWhh[128 + j * 16][32], 64, lane);
            accN[j] = __builtin_amdgcn_wmma_f32_16x16x32_bf16(false, ah1, false, b, (short)0, accN[j], false, false);
        }
#pragma unroll
        for (int j = 0; j < 4; ++j)
#pragma unroll
            for (int v = 0; v < 8; ++v) {
                float r = 1.f / (1.f + expf(-accR[j][v]));
                float z = 1.f / (1.f + expf(-accZ[j][v]));
                float n = tanhf(accI[j][v] + r * accN[j][v]);
                hreg[j][v] = (1.f - z) * n + z * hreg[j][v];
            }
        __syncthreads();
#pragma unroll
        for (int j = 0; j < 4; ++j)
#pragma unroll
            for (int v = 0; v < 8; ++v)
                sH[mb + v][j * 16 + nl] = f2bf(hreg[j][v]);
        __syncthreads();
    }
#pragma unroll
    for (int j = 0; j < 4; ++j)
#pragma unroll
        for (int v = 0; v < 8; ++v)
            seq[(size_t)(n0 + mb + v) * HG + j * 16 + nl] = hreg[j][v];
}

// ---------------- GAT1 projection + attention pre-dots (WMMA) ----------------
__global__ __launch_bounds__(32) void gat1_proj_kernel(
    const float* __restrict__ seq, const float* __restrict__ W,
    const float* __restrict__ asrc, const float* __restrict__ adst,
    float* __restrict__ h1, float* __restrict__ sv, float* __restrict__ dv)
{
    __shared__ unsigned short sW[64][64];   // transposed [n][k]
    __shared__ unsigned short sA[16][64];
    __shared__ float sH[16][64];
    int lane = threadIdx.x;
    int n0 = blockIdx.x * 16;
    for (int idx = lane; idx < 64 * 64; idx += 32) { int n = idx >> 6, k = idx & 63; sW[n][k] = f2bf(W[(size_t)k * 64 + n]); }
    for (int idx = lane; idx < 16 * 64; idx += 32) { int m = idx >> 6, k = idx & 63; sA[m][k] = f2bf(seq[(size_t)(n0 + m) * 64 + k]); }
    __syncthreads();
    v16bf a0 = load_frag16(&sA[0][0], 64, lane);
    v16bf a1 = load_frag16(&sA[0][32], 64, lane);
    int mb = (lane >> 4) * 8, nl = lane & 15;
#pragma unroll
    for (int j = 0; j < 4; ++j) {
        v8f acc;
#pragma unroll
        for (int v = 0; v < 8; ++v) acc[v] = 0.f;
        v16bf b0 = load_frag16(&sW[j * 16][0], 64, lane);
        v16bf b1 = load_frag16(&sW[j * 16][32], 64, lane);
        acc = __builtin_amdgcn_wmma_f32_16x16x32_bf16(false, a0, false, b0, (short)0, acc, false, false);
        acc = __builtin_amdgcn_wmma_f32_16x16x32_bf16(false, a1, false, b1, (short)0, acc, false, false);
#pragma unroll
        for (int v = 0; v < 8; ++v) {
            sH[mb + v][j * 16 + nl] = acc[v];
            h1[(size_t)(n0 + mb + v) * 64 + j * 16 + nl] = acc[v];
        }
    }
    __syncthreads();
    if (lane < 16) {
        float s = 0.f;
        for (int c = 0; c < 64; ++c) s += sH[lane][c] * asrc[c];
        sv[n0 + lane] = s;
    } else {
        int m = lane - 16;
        float s = 0.f;
        for (int c = 0; c < 64; ++c) s += sH[m][c] * adst[c];
        dv[n0 + m] = s;
    }
}

// ---------------- GAT1 edge softmax (atomics) ----------------
__global__ void edge_max_kernel(const int* __restrict__ src, const int* __restrict__ dst, int E,
                                const float* __restrict__ sv, const float* __restrict__ dv,
                                unsigned* __restrict__ maxk) {
    int ET = E + N_COMPANY;
    for (int e = blockIdx.x * blockDim.x + threadIdx.x; e < ET; e += gridDim.x * blockDim.x) {
        int si, di;
        if (e < E) { si = src[e]; di = dst[e]; } else { si = di = e - E; }
        float x = sv[si] + dv[di];
        float ev = x > 0.f ? x : 0.2f * x;
        atomicMax(&maxk[di], fflip(ev));
    }
}

__global__ void edge_exp_kernel(const int* __restrict__ src, const int* __restrict__ dst, int E,
                                const float* __restrict__ sv, const float* __restrict__ dv,
                                const unsigned* __restrict__ maxk,
                                float* __restrict__ exv, float* __restrict__ denom) {
    int ET = E + N_COMPANY;
    for (int e = blockIdx.x * blockDim.x + threadIdx.x; e < ET; e += gridDim.x * blockDim.x) {
        int si, di;
        if (e < E) { si = src[e]; di = dst[e]; } else { si = di = e - E; }
        float x = sv[si] + dv[di];
        float ev = x > 0.f ? x : 0.2f * x;
        float ex = expf(ev - funflip(maxk[di]));
        exv[e] = ex;
        atomicAdd(&denom[di], ex);
    }
}

// 16 threads per edge, 4 columns per thread (float4 load of h1, 4 f32 atomics)
__global__ void edge_agg_kernel(const int* __restrict__ src, const int* __restrict__ dst, int E,
                                const float* __restrict__ exv, const float* __restrict__ denom,
                                const float* __restrict__ h1, float* __restrict__ agg) {
    long long ET = (long long)E + N_COMPANY;
    long long total = ET * 16;
    for (long long i = (long long)blockIdx.x * blockDim.x + threadIdx.x; i < total;
         i += (long long)gridDim.x * blockDim.x) {
        int e  = (int)(i >> 4);
        int c4 = (int)(i & 15) * 4;
        int si, di;
        if (e < E) { si = src[e]; di = dst[e]; } else { si = di = e - E; }
        float alpha = exv[e] / (denom[di] + 1e-16f);
        float4 hv = *(const float4*)(h1 + (size_t)si * 64 + c4);
        float* ap = agg + (size_t)di * 64 + c4;
        atomicAdd(ap + 0, alpha * hv.x);
        atomicAdd(ap + 1, alpha * hv.y);
        atomicAdd(ap + 2, alpha * hv.z);
        atomicAdd(ap + 3, alpha * hv.w);
    }
}

// ---------------- Sector max-pool ----------------
__global__ void secpool_kernel(const float* __restrict__ agg, const float* __restrict__ b1,
                               float* __restrict__ pool) {
    int s = blockIdx.x, c = threadIdx.x;
    float m = -INFINITY;
    for (int i = 0; i < PER_SECTOR; ++i)
        m = fmaxf(m, agg[(size_t)(s * PER_SECTOR + i) * 64 + c]);
    pool[s * 64 + c] = m + b1[c];   // intra_emb = agg + b; max distributes over +b
}

// ---------------- GAT2 (64 sectors, dense: all-pairs + self loops) ----------------
__global__ void gat2_kernel(const float* __restrict__ pool, const float* __restrict__ W,
                            const float* __restrict__ asrc, const float* __restrict__ adst,
                            const float* __restrict__ b2, float* __restrict__ sec_emb) {
    __shared__ float h2[64][64];
    __shared__ float s2[64], d2[64];
    int j = threadIdx.x;
    for (int c = 0; c < 64; ++c) {
        float acc = 0.f;
        for (int k = 0; k < 64; ++k) acc += pool[j * 64 + k] * W[k * 64 + c];
        h2[j][c] = acc;
    }
    float ss = 0.f, dd = 0.f;
    for (int c = 0; c < 64; ++c) { ss += h2[j][c] * asrc[c]; dd += h2[j][c] * adst[c]; }
    s2[j] = ss; d2[j] = dd;
    __syncthreads();
    float mv = -INFINITY;
    for (int i = 0; i < 64; ++i) {
        float x = s2[i] + d2[j];
        mv = fmaxf(mv, x > 0.f ? x : 0.2f * x);
    }
    float al[64];
    float den = 0.f;
    for (int i = 0; i < 64; ++i) {
        float x = s2[i] + d2[j];
        float ev = x > 0.f ? x : 0.2f * x;
        al[i] = expf(ev - mv);
        den += al[i];
    }
    float inv = 1.f / (den + 1e-16f);
    for (int c = 0; c < 64; ++c) {
        float acc = 0.f;
        for (int i = 0; i < 64; ++i) acc += al[i] * h2[i][c];
        sec_emb[j * 64 + c] = acc * inv + b2[c];
    }
}

// ---------------- Fusion MLP + logits + softmax/cumsum/clip (WMMA) ----------------
__global__ __launch_bounds__(32) void fusion_kernel(
    const float* __restrict__ seq, const float* __restrict__ sec_emb,
    const int* __restrict__ sector_ids, const float* __restrict__ agg,
    const float* __restrict__ b1, const float* __restrict__ fW, const float* __restrict__ fb,
    const float* __restrict__ lW, const float* __restrict__ lb, float* __restrict__ out)
{
    __shared__ unsigned short sW[64][192];  // fusion_W transposed [n][k]
    __shared__ unsigned short sF[16][192];  // concat features bf16
    __shared__ float sFU[16][64];
    int lane = threadIdx.x;
    int n0 = blockIdx.x * 16;
    for (int idx = lane; idx < 64 * 192; idx += 32) {
        int n = idx / 192, k = idx % 192;
        sW[n][k] = f2bf(fW[(size_t)k * 64 + n]);
    }
    for (int idx = lane; idx < 16 * 192; idx += 32) {
        int m = idx / 192, k = idx % 192;
        int node = n0 + m;
        float v;
        if (k < 64)       v = seq[(size_t)node * 64 + k];
        else if (k < 128) v = sec_emb[(size_t)sector_ids[node] * 64 + (k - 64)];
        else              v = agg[(size_t)node * 64 + (k - 128)] + b1[k - 128];
        sF[m][k] = f2bf(v);
    }
    __syncthreads();
    int mb = (lane >> 4) * 8, nl = lane & 15;
    v16bf af[6];
#pragma unroll
    for (int kk = 0; kk < 6; ++kk) af[kk] = load_frag16(&sF[0][kk * 32], 192, lane);
#pragma unroll
    for (int j = 0; j < 4; ++j) {
        v8f acc;
        float bb = fb[j * 16 + nl];
#pragma unroll
        for (int v = 0; v < 8; ++v) acc[v] = bb;
#pragma unroll
        for (int kk = 0; kk < 6; ++kk) {
            v16bf b = load_frag16(&sW[j * 16][kk * 32], 192, lane);
            acc = __builtin_amdgcn_wmma_f32_16x16x32_bf16(false, af[kk], false, b, (short)0, acc, false, false);
        }
#pragma unroll
        for (int v = 0; v < 8; ++v) sFU[mb + v][j * 16 + nl] = fmaxf(acc[v], 0.f);
    }
    __syncthreads();
    if (lane < 16) {
        int node = n0 + lane;
        float lg[4];
        for (int o = 0; o < 4; ++o) {
            float acc = lb[o];
            for (int c = 0; c < 64; ++c) acc += sFU[lane][c] * lW[c * 4 + o];
            lg[o] = acc;
        }
        float mx = fmaxf(fmaxf(lg[0], lg[1]), fmaxf(lg[2], lg[3]));
        float den = 0.f;
        for (int o = 0; o < 4; ++o) { lg[o] = expf(lg[o] - mx); den += lg[o]; }
        float inv = 1.f / den;
        float cum = 0.f;
        for (int o = 0; o < 4; ++o) {
            cum += lg[o] * inv;
            out[(size_t)node * 4 + o] = fminf(fmaxf(cum, 5e-8f), 1.f - 5e-8f);
        }
    }
}

extern "C" void kernel_launch(void* const* d_in, const int* in_sizes, int n_in,
                              void* d_out, int out_size, void* d_ws, size_t ws_size,
                              hipStream_t stream) {
    (void)n_in; (void)out_size; (void)ws_size;
    const float* daily      = (const float*)d_in[0];
    const int*   inner      = (const int*)d_in[1];
    const int*   sector_ids = (const int*)d_in[3];
    const float* bn_g = (const float*)d_in[4];
    const float* bn_b = (const float*)d_in[5];
    const float* Wih  = (const float*)d_in[6];
    const float* Whh  = (const float*)d_in[7];
    const float* bih  = (const float*)d_in[8];
    const float* bhh  = (const float*)d_in[9];
    const float* h0   = (const float*)d_in[10];
    const float* g1W  = (const float*)d_in[11];
    const float* g1as = (const float*)d_in[12];
    const float* g1ad = (const float*)d_in[13];
    const float* g1b  = (const float*)d_in[14];
    const float* g2W  = (const float*)d_in[15];
    const float* g2as = (const float*)d_in[16];
    const float* g2ad = (const float*)d_in[17];
    const float* g2b  = (const float*)d_in[18];
    const float* fW   = (const float*)d_in[19];
    const float* fb   = (const float*)d_in[20];
    const float* lW   = (const float*)d_in[21];
    const float* lb   = (const float*)d_in[22];

    int E = in_sizes[1] / 2;
    const int* src = inner;
    const int* dst = inner + E;
    int ET = E + N_COMPANY;

    char* ws = (char*)d_ws;
    size_t off = 0;
    auto alloc = [&](size_t bytes) { size_t o = off; off = (off + bytes + 255) & ~(size_t)255; return o; };

    unsigned short* xn  = (unsigned short*)(ws + alloc((size_t)WIN * N_COMPANY * D_IN * 2));
    float* seq  = (float*)(ws + alloc((size_t)N_COMPANY * HG * 4));
    float* h1   = (float*)(ws + alloc((size_t)N_COMPANY * HG * 4));
    float* agg  = (float*)(ws + alloc((size_t)N_COMPANY * HG * 4));
    float* sv   = (float*)(ws + alloc((size_t)N_COMPANY * 4));
    float* dv   = (float*)(ws + alloc((size_t)N_COMPANY * 4));
    unsigned* maxk = (unsigned*)(ws + alloc((size_t)N_COMPANY * 4));
    float* denom = (float*)(ws + alloc((size_t)N_COMPANY * 4));
    float* exv  = (float*)(ws + alloc((size_t)ET * 4));
    float* mu   = (float*)(ws + alloc(FEAT * 4));
    float* rs   = (float*)(ws + alloc(FEAT * 4));
    float* pool = (float*)(ws + alloc((size_t)N_SECTOR * HG * 4));
    float* sece = (float*)(ws + alloc((size_t)N_SECTOR * HG * 4));

    // BatchNorm
    bn_stats_kernel<<<FEAT, 256, 0, stream>>>(daily, mu, rs);
    {
        size_t total = (size_t)WIN * N_COMPANY * D_IN;
        bn_apply_kernel<<<(unsigned)((total + 255) / 256), 256, 0, stream>>>(daily, mu, rs, bn_g, bn_b, xn);
    }
    // Zero accumulators (maxk=0u is below any flipped-float key; denom/agg = 0.0f)
    zero_kernel<<<(N_COMPANY + 255) / 256, 256, 0, stream>>>(maxk, N_COMPANY);
    zero_kernel<<<(N_COMPANY + 255) / 256, 256, 0, stream>>>((unsigned*)denom, N_COMPANY);
    {
        size_t n = (size_t)N_COMPANY * HG;
        zero_kernel<<<(unsigned)((n + 255) / 256), 256, 0, stream>>>((unsigned*)agg, n);
    }
    // GRU
    gru_kernel<<<N_COMPANY / 16, 32, 0, stream>>>(xn, Wih, Whh, bih, bhh, h0, seq);
    // GAT1 projection + attention dots
    gat1_proj_kernel<<<N_COMPANY / 16, 32, 0, stream>>>(seq, g1W, g1as, g1ad, h1, sv, dv);
    // GAT1 edge softmax + aggregation
    edge_max_kernel<<<(ET + 255) / 256, 256, 0, stream>>>(src, dst, E, sv, dv, maxk);
    edge_exp_kernel<<<(ET + 255) / 256, 256, 0, stream>>>(src, dst, E, sv, dv, maxk, exv, denom);
    {
        long long total = (long long)ET * 16;
        edge_agg_kernel<<<(unsigned)((total + 255) / 256), 256, 0, stream>>>(src, dst, E, exv, denom, h1, agg);
    }
    // Sector pooling, GAT2
    secpool_kernel<<<N_SECTOR, 64, 0, stream>>>(agg, g1b, pool);
    gat2_kernel<<<1, 64, 0, stream>>>(pool, g2W, g2as, g2ad, g2b, sece);
    // Fusion + logits
    fusion_kernel<<<N_COMPANY / 16, 32, 0, stream>>>(seq, sece, sector_ids, agg, g1b, fW, fb, lW, lb, (float*)d_out);
}
